// SACCritic_6451040878722
// MI455X (gfx1250) — matrix operand order repaired and checked by reading, not compiled, not run on hardware
//
#include <hip/hip_runtime.h>
#include <hip/hip_bf16.h>
#include <math.h>

// Problem constants
#define BB   64
#define SS   512
#define DDIM 64
#define HH   256
#define AACT 3
#define IINF 13
#define LLAY 2
#define NHH  4
#define HDD  64
#define H2   512
#define H4   1024

typedef __bf16 bf16_t;
typedef __attribute__((ext_vector_type(16))) __bf16 v16bf;
typedef __attribute__((ext_vector_type(8)))  __bf16 v8bf;
typedef __attribute__((ext_vector_type(8)))  float  v8f;

#define HAVE_TDM (__has_builtin(__builtin_amdgcn_tensor_load_to_lds) && \
                  __has_builtin(__builtin_amdgcn_s_wait_tensorcnt))

static __device__ __forceinline__ v16bf make_afrag(v8bf lo, v8bf hi) {
    v16bf r;
#pragma unroll
    for (int i = 0; i < 8; i++) { r[i] = lo[i]; r[i + 8] = hi[i]; }
    return r;
}

static __device__ __forceinline__ v8f wmma_bf16(v16bf a, v16bf b, v8f c) {
    // D = A(16x32 bf16) * B(32x16 bf16) + C(16x16 f32)
    return __builtin_amdgcn_wmma_f32_16x16x32_bf16(false, a, false, b, (short)0, c,
                                                   false, false);
}

#if HAVE_TDM
typedef __attribute__((ext_vector_type(4))) unsigned int u32x4;
typedef __attribute__((ext_vector_type(8))) int          i32x8;
typedef __attribute__((ext_vector_type(4))) int          i32x4;

// TDM: DMA a 16-row x 512-byte tile (row stride S*H bf16) from global to LDS.
// D# per CDNA5 ISA 8.3/8.4: count=1, type=2, data_size=4B, tile 128x16 units,
// tensor_dim0_stride = 65536 (4B units). lds_addr = low 32 bits of flat LDS ptr.
// 6-arg builtin (clang-23 lane): (g0, g1, g2, g3, g4, cpol).
static __device__ __forceinline__ void tdm_load_xtile(const bf16_t* gsrc, bf16_t* ldst) {
    unsigned long long ga = (unsigned long long)(const void*)gsrc;
    unsigned           la = (unsigned)(unsigned long long)(const void*)ldst;
    u32x4 g0;
    g0[0] = 1u;                                             // count=1, user desc
    g0[1] = la;                                             // lds_addr [63:32]
    g0[2] = (unsigned)(ga & 0xffffffffull);                 // global_addr lo
    g0[3] = (unsigned)((ga >> 32) & 0x1ffffffull) | (2u << 30);  // ga hi | type=2
    i32x8 g1;
    g1[0] = (int)(2u << 16);      // workgroup_mask=0 | data_size=4B (code 2)
    g1[1] = (int)(128u << 16);    // tensor_dim0[15:0]=128 at bits 79:64->w1 hi
    g1[2] = (int)(16u << 16);     // tensor_dim0 hi=0 | tensor_dim1=16
    g1[3] = (int)(128u << 16);    // tensor_dim1 hi=0 | tile_dim0=128
    g1[4] = 16;                   // tile_dim1=16 | tile_dim2=0
    g1[5] = 65536;                // tensor_dim0_stride lo (4B units)
    g1[6] = 0;                    // stride hi | tensor_dim1_stride lo
    g1[7] = 0;
    i32x4 z4 = {0, 0, 0, 0};
    i32x8 z8 = {0, 0, 0, 0, 0, 0, 0, 0};
    __builtin_amdgcn_tensor_load_to_lds(g0, g1, z4, z4, z8, 0);
}
#endif

// ---------------------------------------------------------------------------
// Conversions
// ---------------------------------------------------------------------------
__global__ void k_f32_to_bf16(const float* __restrict__ src, bf16_t* __restrict__ dst,
                              long n) {
    long i = (long)blockIdx.x * blockDim.x + threadIdx.x;
    if (i < n) dst[i] = (bf16_t)src[i];
}

// src [K,N] f32 -> dst [N,K] bf16 (transposed weight for B-fragment loads)
__global__ void k_transpose_bf16(const float* __restrict__ src, bf16_t* __restrict__ dst,
                                 int K, int N) {
    long i = (long)blockIdx.x * blockDim.x + threadIdx.x;
    if (i >= (long)K * N) return;
    int k = (int)(i / N), n = (int)(i % N);
    dst[(long)n * K + k] = (bf16_t)src[i];
}

// gbuf [B,S,NH*HD] f32 -> dst [B,NH,S,HD] bf16 (optionally scaled; for q/k)
__global__ void k_split_heads(const float* __restrict__ src, bf16_t* __restrict__ dst,
                              float scale) {
    long idx = (long)blockIdx.x * blockDim.x + threadIdx.x;
    if (idx >= (long)BB * SS * HH) return;
    int d = (int)(idx % HDD);
    long t = idx / HDD;
    int h = (int)(t % NHH); t /= NHH;
    int s = (int)(t % SS);
    int b = (int)(t / SS);
    dst[(((long)b * NHH + h) * SS + s) * HDD + d] = (bf16_t)(src[idx] * scale);
}

// gbuf [B,S,NH*HD] f32 -> dst [B,NH,HD,S] bf16 (V transposed for B-fragments)
__global__ void k_split_vT(const float* __restrict__ src, bf16_t* __restrict__ dst) {
    long idx = (long)blockIdx.x * blockDim.x + threadIdx.x;
    if (idx >= (long)BB * SS * HH) return;
    int d = (int)(idx % HDD);
    long t = idx / HDD;
    int h = (int)(t % NHH); t /= NHH;
    int s = (int)(t % SS);
    int b = (int)(t / SS);
    dst[(((long)b * NHH + h) * HDD + d) * SS + s] = (bf16_t)src[idx];
}

// ---------------------------------------------------------------------------
// Generic WMMA GEMM: C[M,N] = A[M,K](bf16) * BT[N,K]^T(bf16) + bias
// block = 128 threads (4 waves), wave -> one 16x16 tile; grid = (N/64, M/16)
// ---------------------------------------------------------------------------
__global__ __launch_bounds__(128) void k_gemm_bf16(
    const bf16_t* __restrict__ A, const bf16_t* __restrict__ BT,
    const float* __restrict__ bias, float* __restrict__ C,
    int N, int K, int lda, int ldbt, int ldc) {
    const int lane = threadIdx.x & 31;
    const int wave = threadIdx.x >> 5;
    const int n0 = (blockIdx.x * 4 + wave) * 16;
    const int m0 = blockIdx.y * 16;
    if (n0 >= N) return;
    const int col = lane & 15;
    const int hi  = lane >> 4;
    const bf16_t* arow = A + (long)(m0 + col) * lda;
    const bf16_t* brow = BT + (long)(n0 + col) * ldbt + hi * 16;
    v8f acc = {};
    for (int k0 = 0; k0 < K; k0 += 32) {
        const bf16_t* pa = arow + k0 + hi * 16;
        v16bf a = make_afrag(*(const v8bf*)pa, *(const v8bf*)(pa + 8));
        v16bf b = *(const v16bf*)(brow + k0);
        if (k0 + 32 < K) __builtin_prefetch(brow + k0 + 64, 0, 1);  // global_prefetch_b8
        acc = wmma_bf16(a, b, acc);
    }
    const float bv = bias ? bias[n0 + col] : 0.0f;
    float* crow = C + (long)(m0 + hi * 8) * ldc + n0 + col;
#pragma unroll
    for (int r = 0; r < 8; r++) crow[(long)r * ldc] = acc[r] + bv;
}

// ---------------------------------------------------------------------------
// LayerNorm over rows of width W (W<=512). One wave per row. Optional residual
// add, optional exact-GELU, f32 and/or bf16 outputs (either may be null).
// ---------------------------------------------------------------------------
__global__ __launch_bounds__(32) void k_layernorm(
    const float* __restrict__ in, const float* __restrict__ resid,
    const float* __restrict__ g, const float* __restrict__ bta,
    float* __restrict__ outf, bf16_t* __restrict__ outb, int W, int do_gelu) {
    const long row = blockIdx.x;
    const int lane = threadIdx.x;
    const float* p  = in + row * W;
    const float* rp = resid ? resid + row * W : nullptr;
    float vals[16];
    int cnt = 0;
    float s = 0.f;
    for (int i = lane; i < W; i += 32) {
        float v = p[i] + (rp ? rp[i] : 0.f);
        vals[cnt++] = v;
        s += v;
    }
#pragma unroll
    for (int off = 16; off; off >>= 1) s += __shfl_xor(s, off, 32);
    const float mean = s / W;
    float vs = 0.f;
    for (int c = 0; c < cnt; c++) { float d = vals[c] - mean; vs += d * d; }
#pragma unroll
    for (int off = 16; off; off >>= 1) vs += __shfl_xor(vs, off, 32);
    const float rstd = rsqrtf(vs / W + 1e-5f);
    cnt = 0;
    for (int i = lane; i < W; i += 32) {
        float v = (vals[cnt++] - mean) * rstd * g[i] + bta[i];
        if (do_gelu) v = 0.5f * v * (1.f + erff(v * 0.70710678118f));
        if (outf) outf[row * W + i] = v;
        if (outb) outb[row * W + i] = (bf16_t)v;
    }
}

// ---------------------------------------------------------------------------
// sLSTM scan. Recurrence independent per batch row: grid = 4 blocks of 16 rows.
// block = 256 threads (8 waves); wave w owns output cols [w*32, w*32+32) of H.
// h kept in LDS (bf16) for the h@Wh WMMA; c,n in registers per lane.
// x_t tile (16x256 bf16 = 8KB) double-buffered in LDS via TDM tensor_load_to_lds,
// issued one step ahead by wave 0 and waited with s_wait_tensorcnt, so the only
// global traffic inside the dependent chain is the (L2-resident) weight stream.
// ---------------------------------------------------------------------------
__global__ __launch_bounds__(256) void k_lstm_scan(
    const bf16_t* __restrict__ xn,   // [B,S,H] bf16
    const bf16_t* __restrict__ WT,   // [4H, 2H] bf16
    const float*  __restrict__ bias, // [4H]
    float* __restrict__ cur,         // [B,S,H] f32 (residual add in place)
    float* __restrict__ out,         // d_out (final states)
    int layer) {
    __shared__ bf16_t hlds[16][HH];      // 8 KB
#if HAVE_TDM
    __shared__ bf16_t xlds[2][16][HH];   // 16 KB double buffer for x_t tiles
#endif
    const int lane = threadIdx.x & 31;
    const int wave = threadIdx.x >> 5;
    const int col  = lane & 15;
    const int hi   = lane >> 4;
    const int mb   = blockIdx.x;     // batch tile (16 rows)
    const int j0   = wave * 32;

    for (int i = threadIdx.x; i < 16 * HH; i += 256) ((bf16_t*)hlds)[i] = (bf16_t)0.f;

    float creg[2][8], nreg[2][8], hout[2][8];
#pragma unroll
    for (int s2 = 0; s2 < 2; s2++)
#pragma unroll
        for (int r = 0; r < 8; r++) { creg[s2][r] = 0.f; nreg[s2][r] = 1.f; }

    float bg[2][4];
#pragma unroll
    for (int s2 = 0; s2 < 2; s2++)
#pragma unroll
        for (int gt = 0; gt < 4; gt++) bg[s2][gt] = bias[gt * HH + j0 + s2 * 16 + col];

#if HAVE_TDM
    if (wave == 0)  // one TDM issue per workgroup (wave 0); EXEC ignored by TDM
        tdm_load_xtile(xn + (long)(mb * 16) * SS * HH, &xlds[0][0][0]);
#endif
    __syncthreads();

    for (int t = 0; t < SS; t++) {
#if HAVE_TDM
        if (wave == 0) __builtin_amdgcn_s_wait_tensorcnt(0);  // x_t tile landed
        __syncthreads();                                      // publish to all waves
        if (wave == 0 && t + 1 < SS)                          // prefetch x_{t+1}
            tdm_load_xtile(xn + (long)(mb * 16) * SS * HH + (long)(t + 1) * HH,
                           &xlds[(t + 1) & 1][0][0]);
        const bf16_t* xrow = &xlds[t & 1][col][0];
#else
        const bf16_t* xrow = xn + (((long)(mb * 16 + col) * SS) + t) * HH;
#endif
#pragma unroll
        for (int s2 = 0; s2 < 2; s2++) {
            const int j = j0 + s2 * 16;
            v8f acc0 = {}, acc1 = {}, acc2 = {}, acc3 = {};
            const bf16_t* b0 = WT + ((long)(0 * HH + j + col)) * H2 + hi * 16;
            const bf16_t* b1 = WT + ((long)(1 * HH + j + col)) * H2 + hi * 16;
            const bf16_t* b2 = WT + ((long)(2 * HH + j + col)) * H2 + hi * 16;
            const bf16_t* b3 = WT + ((long)(3 * HH + j + col)) * H2 + hi * 16;
            // x part (K = 0..255): LDS tile (TDM) or global (fallback)
#pragma unroll 2
            for (int k0 = 0; k0 < HH; k0 += 32) {
                const bf16_t* pa = xrow + k0 + hi * 16;
                v16bf a = make_afrag(*(const v8bf*)pa, *(const v8bf*)(pa + 8));
                acc0 = wmma_bf16(a, *(const v16bf*)(b0 + k0), acc0);
                acc1 = wmma_bf16(a, *(const v16bf*)(b1 + k0), acc1);
                acc2 = wmma_bf16(a, *(const v16bf*)(b2 + k0), acc2);
                acc3 = wmma_bf16(a, *(const v16bf*)(b3 + k0), acc3);
            }
            // h part (K = 256..511) from LDS
#pragma unroll 2
            for (int k0 = 0; k0 < HH; k0 += 32) {
                const bf16_t* pa = &hlds[col][k0 + hi * 16];
                v16bf a = make_afrag(*(const v8bf*)pa, *(const v8bf*)(pa + 8));
                const int kk = HH + k0;
                acc0 = wmma_bf16(a, *(const v16bf*)(b0 + kk), acc0);
                acc1 = wmma_bf16(a, *(const v16bf*)(b1 + kk), acc1);
                acc2 = wmma_bf16(a, *(const v16bf*)(b2 + kk), acc2);
                acc3 = wmma_bf16(a, *(const v16bf*)(b3 + kk), acc3);
            }
#pragma unroll
            for (int r = 0; r < 8; r++) {
                float iv = expf(fminf(fmaxf(acc0[r] + bg[s2][0], -4.f), 4.f));
                float fv = expf(fminf(fmaxf(acc1[r] + bg[s2][1], -4.f), 4.f));
                float ov = acc2[r] + bg[s2][2];
                float zv = tanhf(acc3[r] + bg[s2][3]);
                float c  = fminf(fmaxf(fv * creg[s2][r] + iv * zv, -1e6f), 1e6f);
                float nn = fminf(fmaxf(fv * nreg[s2][r] + iv, 1e-6f), 1e6f);
                float h  = (c / nn) / (1.f + expf(-ov));
                if (h != h) h = 0.f;  // nan_to_num
                creg[s2][r] = c; nreg[s2][r] = nn; hout[s2][r] = h;
            }
        }
        __syncthreads();  // all waves done reading old h
#pragma unroll
        for (int s2 = 0; s2 < 2; s2++) {
            const int j = j0 + s2 * 16 + col;
#pragma unroll
            for (int r = 0; r < 8; r++) {
                const int rl = hi * 8 + r;
                const float h = hout[s2][r];
                hlds[rl][j] = (bf16_t)h;
                const long b = mb * 16 + rl;
                cur[((long)b * SS + t) * HH + j] += h;  // residual
            }
        }
        __syncthreads();  // h visible for next step
    }
    // final states -> d_out : [q1(64) q2(64) hs(L*B*H) cs(L*B*H) ns(L*B*H)]
#pragma unroll
    for (int s2 = 0; s2 < 2; s2++) {
        const int j = j0 + s2 * 16 + col;
#pragma unroll
        for (int r = 0; r < 8; r++) {
            const int rl = hi * 8 + r;
            const long b = mb * 16 + rl;
            const long base = (long)layer * BB * HH + b * HH + j;
            out[128 + base] = hout[s2][r];
            out[128 + (long)LLAY * BB * HH + base] = creg[s2][r];
            out[128 + 2L * LLAY * BB * HH + base] = nreg[s2][r];
        }
    }
}

// ---------------------------------------------------------------------------
// Fused attention: one wave per (b, head, 16-query tile).
// scores (f32) + probs (bf16) staged in LDS; both GEMMs via WMMA.
// q pre-scaled by 1/sqrt(HD); v pre-transposed [HD,S].
// ---------------------------------------------------------------------------
__global__ __launch_bounds__(32) void k_attention(
    const bf16_t* __restrict__ q,   // [B,NH,S,HD]
    const bf16_t* __restrict__ k,   // [B,NH,S,HD]
    const bf16_t* __restrict__ vT,  // [B,NH,HD,S]
    bf16_t* __restrict__ o) {       // [B,S,NH*HD] bf16
    __shared__ float  sc[16][SS];    // 32 KB
    __shared__ bf16_t pr[16][SS];    // 16 KB
    const int lane = threadIdx.x;
    const int col = lane & 15, hi = lane >> 4;
    const int q0 = blockIdx.x * 16;
    const int h  = blockIdx.y;
    const int b  = blockIdx.z;
    const bf16_t* qb = q + (((long)b * NHH + h) * SS) * HDD;
    const bf16_t* kb = k + (((long)b * NHH + h) * SS) * HDD;
    const bf16_t* vb = vT + (((long)b * NHH + h) * HDD) * SS;

    // q fragments (K=64 -> 2 chunks), reused across all 32 key tiles
    const bf16_t* qrow = qb + (long)(q0 + col) * HDD;
    v16bf a0 = make_afrag(*(const v8bf*)(qrow + hi * 16), *(const v8bf*)(qrow + 8 + hi * 16));
    v16bf a1 = make_afrag(*(const v8bf*)(qrow + 32 + hi * 16), *(const v8bf*)(qrow + 40 + hi * 16));

    for (int kt = 0; kt < SS / 16; kt++) {
        const bf16_t* krow = kb + (long)(kt * 16 + col) * HDD + hi * 16;
        v8f acc = {};
        acc = wmma_bf16(a0, *(const v16bf*)(krow), acc);
        acc = wmma_bf16(a1, *(const v16bf*)(krow + 32), acc);
#pragma unroll
        for (int r = 0; r < 8; r++) sc[hi * 8 + r][kt * 16 + col] = acc[r];
    }
    __syncthreads();
    {   // softmax: 2 lanes per row, 256 elements each
        const int row = lane & 15;
        const int base = (lane >> 4) * 256;
        float mx = -3.4e38f;
        for (int i = base; i < base + 256; i++) mx = fmaxf(mx, sc[row][i]);
        mx = fmaxf(mx, __shfl_xor(mx, 16, 32));
        float sum = 0.f;
        for (int i = base; i < base + 256; i++) sum += expf(sc[row][i] - mx);
        sum += __shfl_xor(sum, 16, 32);
        const float inv = 1.f / sum;
        for (int i = base; i < base + 256; i++)
            pr[row][i] = (bf16_t)(expf(sc[row][i] - mx) * inv);
    }
    __syncthreads();
    // O[16,64] = P[16,512] @ V[512,64]
    v8f acc[4] = {};
    for (int k0 = 0; k0 < SS; k0 += 32) {
        const bf16_t* pa = &pr[col][k0 + hi * 16];
        v16bf a = make_afrag(*(const v8bf*)pa, *(const v8bf*)(pa + 8));
#pragma unroll
        for (int n = 0; n < 4; n++) {
            const bf16_t* brow = vb + (long)(n * 16 + col) * SS + k0 + hi * 16;
            acc[n] = wmma_bf16(a, *(const v16bf*)brow, acc[n]);
        }
    }
#pragma unroll
    for (int n = 0; n < 4; n++)
#pragma unroll
        for (int r = 0; r < 8; r++) {
            const int s = q0 + hi * 8 + r;
            o[((long)b * SS + s) * HH + h * HDD + n * 16 + col] = (bf16_t)acc[n][r];
        }
}

// ---------------------------------------------------------------------------
// Small dense ops (scalar VALU; tiny K or tiny M)
// ---------------------------------------------------------------------------
__global__ void k_rowdot(const float* __restrict__ in, const float* __restrict__ w,
                         const float* __restrict__ bias, float* __restrict__ out,
                         int M, int N, int K) {
    long idx = (long)blockIdx.x * blockDim.x + threadIdx.x;
    if (idx >= (long)M * N) return;
    int m = (int)(idx / N), n = (int)(idx % N);
    const float* a = in + (long)m * K;
    float s = bias ? bias[n] : 0.f;
    for (int kk = 0; kk < K; kk++) s += a[kk] * w[(long)kk * N + n];
    out[idx] = s;
}

__global__ __launch_bounds__(512) void k_softmax512(const float* __restrict__ in,
                                                    float* __restrict__ out) {
    __shared__ float red[512];
    const int b = blockIdx.x, i = threadIdx.x;
    const float v = in[(long)b * SS + i];
    red[i] = v; __syncthreads();
    for (int off = 256; off; off >>= 1) { if (i < off) red[i] = fmaxf(red[i], red[i + off]); __syncthreads(); }
    const float mx = red[0]; __syncthreads();
    const float e = expf(v - mx);
    red[i] = e; __syncthreads();
    for (int off = 256; off; off >>= 1) { if (i < off) red[i] += red[i + off]; __syncthreads(); }
    out[(long)b * SS + i] = e / red[0];
}

__global__ __launch_bounds__(256) void k_context(const float* __restrict__ xa,
                                                 const float* __restrict__ w,
                                                 float* __restrict__ ctx) {
    const int b = blockIdx.x, j = threadIdx.x;
    float s = 0.f;
    for (int t = 0; t < SS; t++) s += xa[((long)b * SS + t) * HH + j] * w[(long)b * SS + t];
    ctx[(long)b * HH + j] = s;
}

__global__ void k_concat(const float* __restrict__ ctx, const float* __restrict__ ie,
                         const float* __restrict__ act, float* __restrict__ cat) {
    const int W = HH + 64 + AACT;  // 323
    long idx = (long)blockIdx.x * blockDim.x + threadIdx.x;
    if (idx >= (long)BB * W) return;
    int b = (int)(idx / W), j = (int)(idx % W);
    float v;
    if (j < HH)            v = ctx[b * HH + j];
    else if (j < HH + 64)  v = ie[b * 64 + (j - HH)];
    else                   v = act[b * AACT + (j - HH - 64)];
    cat[idx] = v;
}

// ---------------------------------------------------------------------------
extern "C" void kernel_launch(void* const* d_in, const int* in_sizes, int n_in,
                              void* d_out, int out_size, void* d_ws, size_t ws_size,
                              hipStream_t stream) {
    const float* x       = (const float*)d_in[0];
    const float* action  = (const float*)d_in[1];
    const float* info    = (const float*)d_in[2];
    const float* in_w    = (const float*)d_in[3];
    const float* in_b    = (const float*)d_in[4];
    const float* in_ln_g = (const float*)d_in[5];
    const float* in_ln_b = (const float*)d_in[6];
    const float* cell_w  = (const float*)d_in[7];   // [L,512,1024]
    const float* cell_b  = (const float*)d_in[8];   // [L,1024]
    const float* ln_g    = (const float*)d_in[9];   // [L,256]
    const float* ln_b    = (const float*)d_in[10];
    const float* qkv_w   = (const float*)d_in[11];  // [256,768]
    const float* qkv_b   = (const float*)d_in[12];
    const float* out_w   = (const float*)d_in[13];  // [256,256]
    const float* out_b   = (const float*)d_in[14];
    const float* at_ln_g = (const float*)d_in[15];
    const float* at_ln_b = (const float*)d_in[16];
    const float* pool_w  = (const float*)d_in[17];  // [256,1]
    const float* pool_b  = (const float*)d_in[18];
    const float* i_w1    = (const float*)d_in[19];
    const float* i_b1    = (const float*)d_in[20];
    const float* i_l1g   = (const float*)d_in[21];
    const float* i_l1b   = (const float*)d_in[22];
    const float* i_w2    = (const float*)d_in[23];
    const float* i_b2    = (const float*)d_in[24];
    const float* i_l2g   = (const float*)d_in[25];
    const float* i_l2b   = (const float*)d_in[26];
    const float* sa_w1   = (const float*)d_in[27];  // [323,256]
    const float* sa_b1   = (const float*)d_in[28];
    const float* sa_l1g  = (const float*)d_in[29];
    const float* sa_l1b  = (const float*)d_in[30];
    const float* sa_w2   = (const float*)d_in[31];  // [256,128]
    const float* sa_b2   = (const float*)d_in[32];
    const float* sa_l2g  = (const float*)d_in[33];
    const float* sa_l2b  = (const float*)d_in[34];
    const float* q1_w    = (const float*)d_in[35];
    const float* q1_b    = (const float*)d_in[36];
    const float* q2_w    = (const float*)d_in[37];
    const float* q2_b    = (const float*)d_in[38];
    float* out = (float*)d_out;

    // ---- workspace carve ----
    char* wsp = (char*)d_ws;
    auto carve = [&](size_t bytes) {
        void* p = (void*)wsp;
        wsp += (bytes + 255) & ~(size_t)255;
        return p;
    };
    const long MS = (long)BB * SS;  // 32768 rows
    bf16_t* xbf   = (bf16_t*)carve(MS * DDIM * 2);
    bf16_t* in_wT = (bf16_t*)carve((size_t)HH * DDIM * 2);
    bf16_t* cellT = (bf16_t*)carve((size_t)LLAY * H4 * H2 * 2);
    bf16_t* qkvT  = (bf16_t*)carve((size_t)768 * HH * 2);
    bf16_t* outT  = (bf16_t*)carve((size_t)HH * HH * 2);
    float*  cur   = (float*)carve(MS * HH * 4);
    bf16_t* xnbf  = (bf16_t*)carve(MS * HH * 2);  // reused as cur_bf later
    float*  gbuf  = (float*)carve(MS * HH * 4);
    bf16_t* qbf   = (bf16_t*)carve(MS * HH * 2);
    bf16_t* kbf   = (bf16_t*)carve(MS * HH * 2);
    bf16_t* vTbf  = (bf16_t*)carve(MS * HH * 2);
    bf16_t* obf   = (bf16_t*)carve(MS * HH * 2);
    float*  plog  = (float*)carve(MS * 4);
    float*  pwt   = (float*)carve(MS * 4);
    float*  ctx   = (float*)carve((size_t)BB * HH * 4);
    float*  ie1   = (float*)carve((size_t)BB * 64 * 4);
    float*  ie2   = (float*)carve((size_t)BB * 64 * 4);
    float*  cat   = (float*)carve((size_t)BB * 323 * 4);
    float*  f1    = (float*)carve((size_t)BB * 256 * 4);
    float*  f2    = (float*)carve((size_t)BB * 128 * 4);
    (void)ws_size; (void)in_sizes; (void)n_in; (void)out_size;

    auto blocks = [](long n, int bs) { return (unsigned)((n + bs - 1) / bs); };

    // ---- weight preprocessing (bf16 + transpose to [N,K]) ----
    k_f32_to_bf16<<<blocks(MS * DDIM, 256), 256, 0, stream>>>(x, xbf, MS * DDIM);
    k_transpose_bf16<<<blocks((long)DDIM * HH, 256), 256, 0, stream>>>(in_w, in_wT, DDIM, HH);
    for (int l = 0; l < LLAY; l++)
        k_transpose_bf16<<<blocks((long)H2 * H4, 256), 256, 0, stream>>>(
            cell_w + (long)l * H2 * H4, cellT + (long)l * H4 * H2, H2, H4);
    k_transpose_bf16<<<blocks((long)HH * 768, 256), 256, 0, stream>>>(qkv_w, qkvT, HH, 768);
    k_transpose_bf16<<<blocks((long)HH * HH, 256), 256, 0, stream>>>(out_w, outT, HH, HH);

    // ---- input projection + LN ----
    k_gemm_bf16<<<dim3(4, MS / 16), 128, 0, stream>>>(xbf, in_wT, in_b, cur,
                                                      HH, DDIM, DDIM, DDIM, HH);
    k_layernorm<<<MS, 32, 0, stream>>>(cur, nullptr, in_ln_g, in_ln_b, cur, nullptr, HH, 0);

    // ---- sLSTM layers: time-parallel LN -> batch-parallel WMMA scan ----
    for (int l = 0; l < LLAY; l++) {
        k_layernorm<<<MS, 32, 0, stream>>>(cur, nullptr, ln_g + (long)l * HH,
                                           ln_b + (long)l * HH, nullptr, xnbf, HH, 0);
        k_lstm_scan<<<4, 256, 0, stream>>>(xnbf, cellT + (long)l * H4 * H2,
                                           cell_b + (long)l * H4, cur, out, l);
    }

    // ---- attention ----
    k_f32_to_bf16<<<blocks(MS * HH, 256), 256, 0, stream>>>(cur, xnbf, MS * HH);
    const float scale = 0.125f;  // 1/sqrt(64)
    k_gemm_bf16<<<dim3(4, MS / 16), 128, 0, stream>>>(xnbf, qkvT, qkv_b, gbuf,
                                                      HH, HH, HH, HH, HH);
    k_split_heads<<<blocks(MS * HH, 256), 256, 0, stream>>>(gbuf, qbf, scale);
    k_gemm_bf16<<<dim3(4, MS / 16), 128, 0, stream>>>(xnbf, qkvT + (long)256 * HH,
                                                      qkv_b + 256, gbuf, HH, HH, HH, HH, HH);
    k_split_heads<<<blocks(MS * HH, 256), 256, 0, stream>>>(gbuf, kbf, 1.0f);
    k_gemm_bf16<<<dim3(4, MS / 16), 128, 0, stream>>>(xnbf, qkvT + (long)512 * HH,
                                                      qkv_b + 512, gbuf, HH, HH, HH, HH, HH);
    k_split_vT<<<blocks(MS * HH, 256), 256, 0, stream>>>(gbuf, vTbf);
    k_attention<<<dim3(SS / 16, NHH, BB), 32, 0, stream>>>(qbf, kbf, vTbf, obf);
    k_gemm_bf16<<<dim3(4, MS / 16), 128, 0, stream>>>(obf, outT, out_b, gbuf,
                                                      HH, HH, HH, HH, HH);
    // xa = LN(cur + o) -> cur
    k_layernorm<<<MS, 32, 0, stream>>>(gbuf, cur, at_ln_g, at_ln_b, cur, nullptr, HH, 0);

    // ---- attention pooling ----
    k_rowdot<<<blocks(MS, 256), 256, 0, stream>>>(cur, pool_w, pool_b, plog,
                                                  (int)MS, 1, HH);
    k_softmax512<<<BB, 512, 0, stream>>>(plog, pwt);
    k_context<<<BB, 256, 0, stream>>>(cur, pwt, ctx);

    // ---- info encoder ----
    k_rowdot<<<blocks((long)BB * 64, 256), 256, 0, stream>>>(info, i_w1, i_b1, ie1, BB, 64, IINF);
    k_layernorm<<<BB, 32, 0, stream>>>(ie1, nullptr, i_l1g, i_l1b, ie1, nullptr, 64, 1);
    k_rowdot<<<blocks((long)BB * 64, 256), 256, 0, stream>>>(ie1, i_w2, i_b2, ie2, BB, 64, 64);
    k_layernorm<<<BB, 32, 0, stream>>>(ie2, nullptr, i_l2g, i_l2b, ie2, nullptr, 64, 1);

    // ---- heads ----
    k_concat<<<blocks((long)BB * 323, 256), 256, 0, stream>>>(ctx, ie2, action, cat);
    k_rowdot<<<blocks((long)BB * 256, 256), 256, 0, stream>>>(cat, sa_w1, sa_b1, f1, BB, 256, 323);
    k_layernorm<<<BB, 32, 0, stream>>>(f1, nullptr, sa_l1g, sa_l1b, f1, nullptr, 256, 1);
    k_rowdot<<<blocks((long)BB * 128, 256), 256, 0, stream>>>(f1, sa_w2, sa_b2, f2, BB, 128, 256);
    k_layernorm<<<BB, 32, 0, stream>>>(f2, nullptr, sa_l2g, sa_l2b, f2, nullptr, 128, 1);
    k_rowdot<<<blocks(BB, 64), 64, 0, stream>>>(f2, q1_w, q1_b, out, BB, 1, 128);
    k_rowdot<<<blocks(BB, 64), 64, 0, stream>>>(f2, q2_w, q2_b, out + 64, BB, 1, 128);
}
